// EmformerLayer_18485539242030
// MI455X (gfx1250) — matrix operand order repaired and checked by reading, not compiled, not run on hardware
//
#include <hip/hip_runtime.h>
#include <stdint.h>

// ---------- types / helpers ----------
typedef __bf16 bf16x16 __attribute__((ext_vector_type(16)));
typedef float  f32x8  __attribute__((ext_vector_type(8)));
typedef unsigned int u32x4 __attribute__((ext_vector_type(4)));
typedef int i32x8 __attribute__((ext_vector_type(8)));
typedef int i32x4 __attribute__((ext_vector_type(4)));

struct U4x2 { uint4 lo, hi; };

static __device__ __forceinline__ bf16x16 make_frag(uint4 lo, uint4 hi) {
    U4x2 t; t.lo = lo; t.hi = hi;
    return __builtin_bit_cast(bf16x16, t);
}

static __device__ __forceinline__ f32x8 wmma_bf16(bf16x16 a, bf16x16 b, f32x8 c) {
    // D = A x B + C, 16x16x32, bf16 in / f32 acc
    return __builtin_amdgcn_wmma_f32_16x16x32_bf16(false, a, false, b, (short)0, c, false, false);
}

static __device__ __forceinline__ unsigned short f2bf(float f) {
    unsigned int u = __float_as_uint(f);
    u += 0x7FFFu + ((u >> 16) & 1u);   // round-to-nearest-even
    return (unsigned short)(u >> 16);
}

#define NEG_INF (-100000000.0f)

// Shapes: U=1024 B=8 D=512 NHEAD=8 HD=64 R=64 S=32 M=32 Q=1120 KV=1120

// ---------- pack kernels: gather concat + f32 -> bf16 ----------
__global__ void pack_qin_kernel(const float* __restrict__ rc, const float* __restrict__ ut,
                                const float* __restrict__ sm, unsigned short* __restrict__ dst)
{
    int i = blockIdx.x * blockDim.x + threadIdx.x;     // 1120*8*128 groups of 4
    if (i >= 1120 * 8 * 128) return;
    int l = i / (8 * 128);
    int r = i % (8 * 128);
    int b = r / 128;
    int d = (r % 128) * 4;
    const float* src;
    if (l < 64)        src = rc + ((size_t)(l * 8 + b)) * 512 + d;
    else if (l < 1088) src = ut + ((size_t)((l - 64) * 8 + b)) * 512 + d;
    else               src = sm + ((size_t)((l - 1088) * 8 + b)) * 512 + d;
    float4 f = *(const float4*)src;
    uint2 u;
    u.x = (unsigned)f2bf(f.x) | ((unsigned)f2bf(f.y) << 16);
    u.y = (unsigned)f2bf(f.z) | ((unsigned)f2bf(f.w) << 16);
    *(uint2*)(dst + (size_t)i * 4) = u;
}

__global__ void pack_kvin_kernel(const float* __restrict__ mem, const float* __restrict__ rc,
                                 const float* __restrict__ ut, unsigned short* __restrict__ dst)
{
    int i = blockIdx.x * blockDim.x + threadIdx.x;
    if (i >= 1120 * 8 * 128) return;
    int l = i / (8 * 128);
    int r = i % (8 * 128);
    int b = r / 128;
    int d = (r % 128) * 4;
    const float* src;
    if (l < 32)       src = mem + ((size_t)(l * 8 + b)) * 512 + d;
    else if (l < 96)  src = rc + ((size_t)((l - 32) * 8 + b)) * 512 + d;
    else              src = ut + ((size_t)((l - 96) * 8 + b)) * 512 + d;
    float4 f = *(const float4*)src;
    uint2 u;
    u.x = (unsigned)f2bf(f.x) | ((unsigned)f2bf(f.y) << 16);
    u.y = (unsigned)f2bf(f.z) | ((unsigned)f2bf(f.w) << 16);
    *(uint2*)(dst + (size_t)i * 4) = u;
}

__global__ void cvt_bf16_kernel(const float* __restrict__ src, unsigned short* __restrict__ dst, int n)
{
    int i = blockIdx.x * blockDim.x + threadIdx.x;
    if (i * 4 >= n) return;
    float4 f = *(const float4*)(src + (size_t)i * 4);
    uint2 u;
    u.x = (unsigned)f2bf(f.x) | ((unsigned)f2bf(f.y) << 16);
    u.y = (unsigned)f2bf(f.z) | ((unsigned)f2bf(f.w) << 16);
    *(uint2*)(dst + (size_t)i * 4) = u;
}

// ---------- WMMA GEMM:  C[M,N] = A[M,512] * W[N,512]^T + bias  ----------
// W panel (64 cols x 512 K, bf16) staged once in LDS; A software-pipelined from global.
// mode 0: query  -> scale 0.125, scatter to Qh[(b*8+h)][l][e] (bf16)
// mode 1: kv     -> n<512 -> Kh, n>=512 -> Vh (bf16)
// mode 2: output -> f32 to d_out with rcu / clipped-mem split
__global__ __launch_bounds__(256)
void gemm_bf16_kernel(const unsigned short* __restrict__ A,
                      const unsigned short* __restrict__ W,
                      const float* __restrict__ bias,
                      int mode,
                      unsigned short* __restrict__ dst0,
                      unsigned short* __restrict__ dst1,
                      float* __restrict__ dstf)
{
    __shared__ __align__(16) unsigned short Bp[64 * 520];   // [col][k], padded row 520

    const int tid  = threadIdx.x;
    const int lane = tid & 31;
    const int wid  = tid >> 5;                  // 8 waves
    const int nl   = lane & 15;
    const int kh   = lane >> 4;                 // 0/1
    const int row0 = blockIdx.y * 128 + wid * 16;
    const int col0 = blockIdx.x * 64;
    const int arow = row0 + nl;

    // cooperative stage of the W panel (shared by all 8 waves)
    for (int i = tid; i < 4096; i += 256) {
        int c  = i >> 6;            // 0..63
        int kg = (i & 63) * 8;      // 0..504
        uint4 w = *(const uint4*)(W + (size_t)(col0 + c) * 512 + kg);
        *(uint4*)(&Bp[c * 520 + kg]) = w;
    }
    __syncthreads();

    f32x8 z = {0.f,0.f,0.f,0.f,0.f,0.f,0.f,0.f};
    f32x8 acc[4] = { z, z, z, z };

    const unsigned short* aptr = A + (size_t)arow * 512;
    uint4 alo = *(const uint4*)(aptr + kh * 8);
    uint4 ahi = *(const uint4*)(aptr + 16 + kh * 8);
    for (int kb = 0; kb < 512; kb += 32) {
        uint4 nlo = alo, nhi = ahi;
        if (kb + 32 < 512) {        // software pipeline: next A fragment in flight
            nlo = *(const uint4*)(aptr + kb + 32 + kh * 8);
            nhi = *(const uint4*)(aptr + kb + 48 + kh * 8);
        }
        bf16x16 af = make_frag(alo, ahi);
#pragma unroll
        for (int t = 0; t < 4; ++t) {
            const unsigned short* bptr = &Bp[(t * 16 + nl) * 520 + kb + kh * 8];
            bf16x16 bfr = make_frag(*(const uint4*)(bptr), *(const uint4*)(bptr + 16));
            acc[t] = wmma_bf16(af, bfr, acc[t]);
        }
        alo = nlo; ahi = nhi;
    }

#pragma unroll
    for (int t = 0; t < 4; ++t) {
        int ncol = col0 + t * 16 + nl;
        float bv = bias[ncol];
#pragma unroll
        for (int v = 0; v < 8; ++v) {
            int mrow = row0 + v + 8 * kh;       // C layout: M = v (+8 for hi half)
            float val = acc[t][v] + bv;
            int l = mrow >> 3, b = mrow & 7;
            if (mode == 0) {
                val *= 0.125f;                  // hd^-0.5
                int h = ncol >> 6, e = ncol & 63;
                dst0[(((size_t)(b * 8 + h)) * 1120 + l) * 64 + e] = f2bf(val);
            } else if (mode == 1) {
                if (ncol < 512) {
                    int h = ncol >> 6, e = ncol & 63;
                    dst0[(((size_t)(b * 8 + h)) * 1120 + l) * 64 + e] = f2bf(val);
                } else {
                    int n2 = ncol - 512;
                    int h = n2 >> 6, e = n2 & 63;
                    dst1[(((size_t)(b * 8 + h)) * 1120 + l) * 64 + e] = f2bf(val);
                }
            } else {
                if (l < 1088) {
                    dstf[((size_t)(l * 8 + b)) * 512 + ncol] = val;
                } else if (l < 1119) {
                    float cv = fminf(fmaxf(val, -10.0f), 10.0f);
                    dstf[(size_t)1088 * 8 * 512 + ((size_t)((l - 1088) * 8 + b)) * 512 + ncol] = cv;
                } // l == 1119 dropped (out_mem[:-1])
            }
        }
    }
}

// ---------- flash attention: 64 heads, q_len=kv_len=1120, hd=64 ----------
// block: 224 threads = 7 waves x 16 q rows = 112 q rows ; grid (10, 64)
// K chunk staged via the Tensor Data Mover (TDM) with padded-LDS descriptor.
__global__ __launch_bounds__(224)
void attn_kernel(const unsigned short* __restrict__ Qh,
                 const unsigned short* __restrict__ Kh,
                 const unsigned short* __restrict__ Vh,
                 const unsigned char* __restrict__ amask,
                 const int* __restrict__ lengths,
                 unsigned short* __restrict__ att)
{
    __shared__ __align__(16) unsigned short K_lds[32 * 72];       // [kv][d], row 72 via TDM pad
    __shared__ __align__(16) unsigned short Vt_lds[64 * 40];      // [d][kv], padded row 40
    __shared__ __align__(16) unsigned short P_lds[7 * 16 * 32];   // per-wave P tile

    const int tid  = threadIdx.x;
    const int lane = tid & 31;
    const int wid  = tid >> 5;                 // 0..6
    const int nl   = lane & 15;
    const int kh   = lane >> 4;
    const int head = blockIdx.y;               // b*8 + h
    const int bidx = head >> 3;
    const int hidx = head & 7;
    const int q0   = blockIdx.x * 112 + wid * 16;
    const int thresh = 96 + lengths[bidx];     // kv_len - u_len + lengths[b]

    // Q fragments (already scaled by 0.125 in GEMM epilogue)
    const unsigned short* qptr = Qh + ((size_t)head * 1120 + q0 + nl) * 64;
    bf16x16 aq0 = make_frag(*(const uint4*)(qptr + kh * 8), *(const uint4*)(qptr + 16 + kh * 8));
    bf16x16 aq1 = make_frag(*(const uint4*)(qptr + 32 + kh * 8), *(const uint4*)(qptr + 48 + kh * 8));

    f32x8 z = {0.f,0.f,0.f,0.f,0.f,0.f,0.f,0.f};
    f32x8 acc[4] = { z, z, z, z };
    float mrow[8], lrow[8];
#pragma unroll
    for (int v = 0; v < 8; ++v) { mrow[v] = NEG_INF; lrow[v] = 0.f; }

    unsigned short* pw = &P_lds[wid * 16 * 32];

    // TDM descriptor (D#) group1: data_size=2B, pad_enable, pad after 32 DWORDs (code 4)
    // by 4 DWORDs (code 3) -> 72-half LDS rows. tensor_dim0=stride0=64, tile 64x32.
    const unsigned kldsa = (unsigned)(uintptr_t)(&K_lds[0]);   // LDS addr = addr[31:0]
    i32x8 g1;
    g1[0] = (1 << 16) | (1 << 20) | (4 << 22) | (3 << 25);
    g1[1] = (64 << 16);        // tensor_dim0 = 64       (bits 79:48)
    g1[2] = (32 << 16);        // tensor_dim1 = 32       (bits 111:80)
    g1[3] = (64 << 16);        // tile_dim0   = 64       (bits 127:112)
    g1[4] = 32;                // tile_dim1   = 32, tile_dim2 = 0
    g1[5] = 64;                // tensor_dim0_stride = 64 (bits 207:160)
    g1[6] = 0;
    g1[7] = 0;
    i32x4 gz = {0, 0, 0, 0};

    for (int c = 0; c < 35; ++c) {
        const int kv0 = c * 32;
        __syncthreads();       // previous chunk fully consumed

        // ---- K chunk: TDM async DMA into padded LDS (wave 0 issues) ----
        if (wid == 0) {
            unsigned long long ga =
                (unsigned long long)(uintptr_t)(Kh + ((size_t)head * 1120 + kv0) * 64);
            u32x4 g0;
            g0.x = 1u;                                   // count = 1
            g0.y = kldsa;                                // lds_addr
            g0.z = (unsigned)(ga & 0xFFFFFFFFu);         // global_addr[31:0]
            g0.w = (unsigned)((ga >> 32) & 0x01FFFFFFu) | (2u << 30);  // [56:32] | type=2
#if defined(__clang_major__) && __clang_major__ >= 23
            i32x8 gz8 = {0, 0, 0, 0, 0, 0, 0, 0};
            __builtin_amdgcn_tensor_load_to_lds(g0, g1, gz, gz, gz8, 0);
#else
            __builtin_amdgcn_tensor_load_to_lds(g0, g1, gz, gz, 0);
#endif
        }

        // ---- V chunk: cooperative transpose-stage (TDM cannot transpose) ----
        for (int i = tid; i < 256; i += 224) {
            int r = i >> 3, cg = i & 7;
            uint4 vq = *(const uint4*)(Vh + ((size_t)head * 1120 + kv0 + r) * 64 + cg * 8);
            unsigned int w[4] = { vq.x, vq.y, vq.z, vq.w };
#pragma unroll
            for (int j = 0; j < 4; ++j) {
                Vt_lds[(cg * 8 + 2 * j)     * 40 + r] = (unsigned short)(w[j] & 0xFFFFu);
                Vt_lds[(cg * 8 + 2 * j + 1) * 40 + r] = (unsigned short)(w[j] >> 16);
            }
        }
        if (c + 1 < 35) // prefetch next V chunk (global_prefetch_b8)
            __builtin_prefetch(Vh + ((size_t)head * 1120 + kv0 + 32) * 64, 0, 0);

        if (wid == 0) __builtin_amdgcn_s_wait_tensorcnt(0);  // TENSORcnt is per-wave
        __syncthreads();                                     // publish K/V to all waves

        // ---- scores: S = Q * K^T  (two 16x16 kv tiles, K-dim = d = 2x32) ----
        f32x8 s0 = z, s1 = z;
        {
            const unsigned short* p = &K_lds[nl * 72 + kh * 8];
            s0 = wmma_bf16(aq0, make_frag(*(const uint4*)(p), *(const uint4*)(p + 16)), s0);
            p = &K_lds[nl * 72 + 32 + kh * 8];
            s0 = wmma_bf16(aq1, make_frag(*(const uint4*)(p), *(const uint4*)(p + 16)), s0);
            p = &K_lds[(16 + nl) * 72 + kh * 8];
            s1 = wmma_bf16(aq0, make_frag(*(const uint4*)(p), *(const uint4*)(p + 16)), s1);
            p = &K_lds[(16 + nl) * 72 + 32 + kh * 8];
            s1 = wmma_bf16(aq1, make_frag(*(const uint4*)(p), *(const uint4*)(p + 16)), s1);
        }

        // ---- masks + online softmax (row = 16 lanes of a half-wave) ----
        float rescale[8];
#pragma unroll
        for (int v = 0; v < 8; ++v) {
            int qg  = q0 + v + 8 * kh;
            int kg0 = kv0 + nl;
            int kg1 = kv0 + 16 + nl;
            unsigned char am0 = amask[(size_t)qg * 1120 + kg0];
            unsigned char am1 = amask[(size_t)qg * 1120 + kg1];
            float v0 = (am0 || kg0 >= thresh) ? NEG_INF : s0[v];
            float v1 = (am1 || kg1 >= thresh) ? NEG_INF : s1[v];
            float r = fmaxf(v0, v1);
            r = fmaxf(r, __shfl_xor(r, 1, 32));
            r = fmaxf(r, __shfl_xor(r, 2, 32));
            r = fmaxf(r, __shfl_xor(r, 4, 32));
            r = fmaxf(r, __shfl_xor(r, 8, 32));
            float mnew = fmaxf(mrow[v], r);
            float sc = __expf(mrow[v] - mnew);
            float p0 = __expf(v0 - mnew);
            float p1 = __expf(v1 - mnew);
            float rs = p0 + p1;
            rs += __shfl_xor(rs, 1, 32);
            rs += __shfl_xor(rs, 2, 32);
            rs += __shfl_xor(rs, 4, 32);
            rs += __shfl_xor(rs, 8, 32);
            lrow[v] = lrow[v] * sc + rs;
            mrow[v] = mnew;
            rescale[v] = sc;
            // stash P in per-wave LDS for C-layout -> A-layout relayout
            pw[(v + 8 * kh) * 32 + nl]      = f2bf(p0);
            pw[(v + 8 * kh) * 32 + 16 + nl] = f2bf(p1);
        }
#pragma unroll
        for (int t = 0; t < 4; ++t)
#pragma unroll
            for (int v = 0; v < 8; ++v) acc[t][v] *= rescale[v];

        asm volatile("s_wait_dscnt 0" ::: "memory");  // LDS write->read fence (in-wave)

        // P as A fragment (16 q x 32 kv)
        const unsigned short* pp = pw + nl * 32;
        bf16x16 pa = make_frag(*(const uint4*)(pp + kh * 8), *(const uint4*)(pp + 16 + kh * 8));

        // ---- acc += P * V  (4 d-column tiles) ----
#pragma unroll
        for (int t = 0; t < 4; ++t) {
            const unsigned short* vp = &Vt_lds[(t * 16 + nl) * 40 + kh * 8];
            bf16x16 bv = make_frag(*(const uint4*)(vp), *(const uint4*)(vp + 16));
            acc[t] = wmma_bf16(pa, bv, acc[t]);
        }
    }

    // ---- normalize + store att[(q*8+b)*512 + h*64 + d] as bf16 ----
    float rl[8];
#pragma unroll
    for (int v = 0; v < 8; ++v) rl[v] = 1.0f / lrow[v];
#pragma unroll
    for (int t = 0; t < 4; ++t) {
#pragma unroll
        for (int v = 0; v < 8; ++v) {
            int qg = q0 + v + 8 * kh;
            int dc = t * 16 + nl;
            att[((size_t)qg * 8 + bidx) * 512 + hidx * 64 + dc] = f2bf(acc[t][v] * rl[v]);
        }
    }
}

// ---------- host launch ----------
extern "C" void kernel_launch(void* const* d_in, const int* in_sizes, int n_in,
                              void* d_out, int out_size, void* d_ws, size_t ws_size,
                              hipStream_t stream)
{
    (void)in_sizes; (void)n_in; (void)out_size; (void)ws_size;
    const float* utter = (const float*)d_in[0];
    const int*   lens  = (const int*)d_in[1];
    const float* rctx  = (const float*)d_in[2];
    const float* summ  = (const float*)d_in[3];
    const float* memo  = (const float*)d_in[4];
    const unsigned char* amask = (const unsigned char*)d_in[5];
    const float* Wq    = (const float*)d_in[6];
    const float* bq    = (const float*)d_in[7];
    const float* Wkv   = (const float*)d_in[8];
    const float* bkv   = (const float*)d_in[9];
    const float* Wout  = (const float*)d_in[10];
    const float* bout  = (const float*)d_in[11];

    char* ws = (char*)d_ws;
    size_t o = 0;
    auto alloc = [&](size_t bytes) { char* p = ws + o; o += (bytes + 255) & ~(size_t)255; return p; };
    unsigned short* qin   = (unsigned short*)alloc((size_t)8960 * 512 * 2);
    unsigned short* kvin  = (unsigned short*)alloc((size_t)8960 * 512 * 2);
    unsigned short* wqb   = (unsigned short*)alloc((size_t)512 * 512 * 2);
    unsigned short* wkvb  = (unsigned short*)alloc((size_t)1024 * 512 * 2);
    unsigned short* wob   = (unsigned short*)alloc((size_t)512 * 512 * 2);
    unsigned short* qh    = (unsigned short*)alloc((size_t)64 * 1120 * 64 * 2);
    unsigned short* khds  = (unsigned short*)alloc((size_t)64 * 1120 * 64 * 2);
    unsigned short* vhds  = (unsigned short*)alloc((size_t)64 * 1120 * 64 * 2);
    unsigned short* attb  = (unsigned short*)alloc((size_t)8960 * 512 * 2);

    // pack / convert
    pack_qin_kernel<<<4480, 256, 0, stream>>>(rctx, utter, summ, qin);
    pack_kvin_kernel<<<4480, 256, 0, stream>>>(memo, rctx, utter, kvin);
    cvt_bf16_kernel<<<(512 * 512 / 4 + 255) / 256, 256, 0, stream>>>(Wq, wqb, 512 * 512);
    cvt_bf16_kernel<<<(1024 * 512 / 4 + 255) / 256, 256, 0, stream>>>(Wkv, wkvb, 1024 * 512);
    cvt_bf16_kernel<<<(512 * 512 / 4 + 255) / 256, 256, 0, stream>>>(Wout, wob, 512 * 512);

    // query = q_in @ Wq^T + bq, scaled, head-major
    gemm_bf16_kernel<<<dim3(8, 70), 256, 0, stream>>>(qin, wqb, bq, 0, qh, nullptr, nullptr);
    // kv = kv_in @ Wkv^T + bkv, split K/V head-major
    gemm_bf16_kernel<<<dim3(16, 70), 256, 0, stream>>>(kvin, wkvb, bkv, 1, khds, vhds, nullptr);
    // flash attention
    attn_kernel<<<dim3(10, 64), 224, 0, stream>>>(qh, khds, vhds, amask, lens, attb);
    // out = att @ Wout^T + bout, rcu/mem split + clip
    gemm_bf16_kernel<<<dim3(8, 70), 256, 0, stream>>>(attb, wob, bout, 2, nullptr, nullptr, (float*)d_out);
}
